// MaskedSelfAttention_32521492365975
// MI455X (gfx1250) — matrix operand order repaired and checked by reading, not compiled, not run on hardware
//
#include <hip/hip_runtime.h>
#include <hip/hip_bf16.h>

// ---------- constants for this problem ----------
#define BATCH   2
#define SEQ     2048
#define DMODEL  1024
#define HEADS   16
#define HD      64            // head dim
#define MTOT    (BATCH*SEQ)   // 4096 rows in flattened [B*S, D]

// d_out layout (floats): out | attn | K | V
#define ATTN_OFF  ((size_t)MTOT*DMODEL)                            // 4,194,304
#define K_OFF     (ATTN_OFF + (size_t)BATCH*HEADS*SEQ*SEQ)         // +134,217,728
#define V_OFF     (K_OFF + (size_t)MTOT*DMODEL)

typedef __attribute__((ext_vector_type(16))) __bf16 v16bf;
typedef __attribute__((ext_vector_type(8)))  __bf16 v8bf;
typedef __attribute__((ext_vector_type(8)))  float  v8f;
typedef __attribute__((ext_vector_type(4)))  int    v4i;

#define LDS_STRIDE 80   // 160B row stride: multiple of 32B -> aligned 16B/32B frag reads

// ---------- CDNA5 async global->LDS copy (16B per lane), ASYNCcnt tracked ----------
static __device__ __forceinline__ void async_copy16(const void* g, void* l) {
#if __has_builtin(__builtin_amdgcn_global_load_async_to_lds_b128)
  __builtin_amdgcn_global_load_async_to_lds_b128(
      (__attribute__((address_space(1))) v4i*)g,
      (__attribute__((address_space(3))) v4i*)l, 0, 0);
#else
  unsigned      lo = (unsigned)(size_t)l;       // LDS byte address = low 32 bits of flat
  unsigned long long ga = (unsigned long long)(size_t)g;
  asm volatile("global_load_async_to_lds_b128 %0, %1, off"
               :: "v"(lo), "v"(ga) : "memory");
#endif
}
static __device__ __forceinline__ void wait_async() {
#if __has_builtin(__builtin_amdgcn_s_wait_asynccnt)
  __builtin_amdgcn_s_wait_asynccnt(0);
#else
  asm volatile("s_wait_asynccnt 0" ::: "memory");
#endif
}

// A-matrix fragment (16x32 bf16): lane L<16 -> row L, K = {c..c+7, c+16..c+23}, c = hi*8
static __device__ __forceinline__ v16bf frag_a(const __bf16* rowk, int hi) {
  v8bf lo = *(const v8bf*)(rowk + hi*8);
  v8bf hp = *(const v8bf*)(rowk + hi*8 + 16);
  return __builtin_shufflevector(lo, hp, 0,1,2,3,4,5,6,7,8,9,10,11,12,13,14,15);
}
// B-matrix fragment (32x16 bf16), LDS stored [N][K]: lane L<16 -> col L, K = {c..c+7, c+16..c+23}, c = hi*8
static __device__ __forceinline__ v16bf frag_b(const __bf16* rowk, int hi) {
  v8bf lo = *(const v8bf*)(rowk + hi*8);
  v8bf hp = *(const v8bf*)(rowk + hi*8 + 16);
  return __builtin_shufflevector(lo, hp, 0,1,2,3,4,5,6,7,8,9,10,11,12,13,14,15);
}
static __device__ __forceinline__ v8f wmma_bf16(v16bf a, v16bf b, v8f c) {
  return __builtin_amdgcn_wmma_f32_16x16x32_bf16(false, a, false, b, (short)0, c, false, false);
}
static __device__ __forceinline__ float redmax16(float t) {
  t = fmaxf(t, __shfl_xor(t, 1, 16));
  t = fmaxf(t, __shfl_xor(t, 2, 16));
  t = fmaxf(t, __shfl_xor(t, 4, 16));
  t = fmaxf(t, __shfl_xor(t, 8, 16));
  return t;
}
static __device__ __forceinline__ float redsum16(float t) {
  t += __shfl_xor(t, 1, 16);
  t += __shfl_xor(t, 2, 16);
  t += __shfl_xor(t, 4, 16);
  t += __shfl_xor(t, 8, 16);
  return t;
}

// ================= Kernel 1: QKV projection GEMM =================
// C[64x64] tile of (x @ W + b); blockIdx.z selects Q/K/V.
//  z=0: Q -> bf16 stage only (ws)      [f32 Q never needed in output]
//  z=1: K -> f32 output + bf16 stage (ws, natural layout)
//  z=2: V -> f32 output + bf16 stage (ws, per-head TRANSPOSED [b][h][d][s])
__global__ __launch_bounds__(128) void qkv_gemm_kernel(
    const float* __restrict__ x,
    const float* __restrict__ Wq, const float* __restrict__ bq,
    const float* __restrict__ Wk, const float* __restrict__ bk,
    const float* __restrict__ Wv, const float* __restrict__ bv,
    float* __restrict__ Ko, float* __restrict__ Vo,
    __bf16* __restrict__ qb, __bf16* __restrict__ kb, __bf16* __restrict__ vtb)
{
  __shared__ __align__(16) __bf16 Al[64][LDS_STRIDE];  // x tile [M][K]
  __shared__ __align__(16) __bf16 Bt[64][LDS_STRIDE];  // W tile [N][K] (transposed)

  const int z = blockIdx.z;
  const float* W; const float* bias;
  if (z == 0)      { W = Wq; bias = bq; }
  else if (z == 1) { W = Wk; bias = bk; }
  else             { W = Wv; bias = bv; }

  const int tid = threadIdx.x, lane = tid & 31, w = tid >> 5;
  const int hi = lane >> 4, lr = lane & 15;
  const int m0 = blockIdx.x * 64, n0 = blockIdx.y * 64;

  v8f c[4] = {};
  for (int k0 = 0; k0 < DMODEL; k0 += 64) {
    __syncthreads();
    #pragma unroll 4
    for (int i = tid; i < 4096; i += 128) {
      int r = i >> 6, cc = i & 63;
      Al[r][cc] = (__bf16)x[(size_t)(m0 + r) * DMODEL + (k0 + cc)];
      Bt[cc][r] = (__bf16)W[(size_t)(k0 + r) * DMODEL + (n0 + cc)];
    }
    __syncthreads();
    #pragma unroll
    for (int s = 0; s < 2; ++s) {
      v16bf a = frag_a(&Al[w*16 + lr][s*32], hi);
      #pragma unroll
      for (int nd = 0; nd < 4; ++nd) {
        v16bf bb = frag_b(&Bt[nd*16 + lr][s*32], hi);
        c[nd] = wmma_bf16(a, bb, c[nd]);
      }
    }
  }

  #pragma unroll
  for (int nd = 0; nd < 4; ++nd) {
    int n = n0 + nd*16 + lr;
    float bvv = bias[n];
    float vals[8];
    #pragma unroll
    for (int v = 0; v < 8; ++v) vals[v] = c[nd][v] + bvv;

    if (z == 2) {
      #pragma unroll
      for (int v = 0; v < 8; ++v) {
        int row = m0 + w*16 + v + 8*hi;        // C layout: row = v + 8*(lane>=16)
        Vo[(size_t)row * DMODEL + n] = vals[v];
      }
      // packed bf16 transposed stage: V^T[b][h][d][s], 8 consecutive s per lane
      v8bf pk;
      #pragma unroll
      for (int v = 0; v < 8; ++v) pk[v] = (__bf16)vals[v];
      int bb2 = m0 >> 11;                       // batch = m0/2048
      int s0  = (m0 & 2047) + w*16 + 8*hi;      // seq position of vals[0]
      int hh  = n >> 6, d = n & 63;
      *(v8bf*)&vtb[((size_t)(bb2*HEADS + hh) * HD + d) * SEQ + s0] = pk;
    } else {
      __bf16* stage = (z == 0) ? qb : kb;
      #pragma unroll
      for (int v = 0; v < 8; ++v) {
        int row = m0 + w*16 + v + 8*hi;
        stage[(size_t)row * DMODEL + n] = (__bf16)vals[v];
        if (z == 1) Ko[(size_t)row * DMODEL + n] = vals[v];
      }
    }
  }
}

// ================= Kernel 2: causal attention =================
// block = (qt, h, b); 64 query rows; two-pass softmax; tiles arrive via
// GLOBAL_LOAD_ASYNC_TO_LDS_B128 from the bf16 staging buffers.
__global__ __launch_bounds__(128) void attention_kernel(
    const __bf16* __restrict__ qb, const __bf16* __restrict__ kb,
    const __bf16* __restrict__ vtb,
    float* __restrict__ outp, float* __restrict__ attnp)
{
  __shared__ __align__(16) __bf16 Ql[64][LDS_STRIDE];      // Q   [m][d]
  __shared__ __align__(16) __bf16 Kl[64][LDS_STRIDE];      // K   [key][d]  ([N][K] for QK^T)
  __shared__ __align__(16) __bf16 Vt[64][LDS_STRIDE];      // V^T [d][key]  ([N][K] for P.V)
  __shared__ __align__(16) __bf16 Pl[4][16][LDS_STRIDE];   // per-wave P tile [m][key]

  const int tid = threadIdx.x, lane = tid & 31, w = tid >> 5;
  const int hi = lane >> 4, lr = lane & 15;
  const int qt = blockIdx.x, h = blockIdx.y, b = blockIdx.z;
  const size_t rowbase = (size_t)b * SEQ;
  const int q0 = qt * 64;
  const int d0 = h * HD;
  const size_t vt_head = ((size_t)(b*HEADS + h)) * HD * SEQ;  // V^T head base
  const float scale = 0.125f;   // 1/sqrt(64)

  // ---- async load Q tile (bf16, natural) : 512 x 16B chunks ----
  for (int i = tid; i < 512; i += 128) {
    int r = i >> 3, cch = i & 7;
    async_copy16(qb + (rowbase + q0 + r) * DMODEL + d0 + cch*8, &Ql[r][cch*8]);
  }
  wait_async();
  __syncthreads();

  v16bf aq[2];
  aq[0] = frag_a(&Ql[w*16 + lr][0], hi);
  aq[1] = frag_a(&Ql[w*16 + lr][32], hi);

  float mrow[8], lrow[8];
  #pragma unroll
  for (int v = 0; v < 8; ++v) { mrow[v] = -__builtin_inff(); lrow[v] = 0.0f; }

  // ---- pass 1: row max / row sum (online) ----
  for (int kt = 0; kt <= qt; ++kt) {
    __syncthreads();
    for (int i = tid; i < 512; i += 128) {
      int r = i >> 3, cch = i & 7;
      async_copy16(kb + (rowbase + (size_t)kt*64 + r) * DMODEL + d0 + cch*8,
                   &Kl[r][cch*8]);
    }
    wait_async();
    __syncthreads();

    v8f cs[4] = {};
    #pragma unroll
    for (int s = 0; s < 2; ++s) {
      #pragma unroll
      for (int nd = 0; nd < 4; ++nd) {
        v16bf bk = frag_b(&Kl[nd*16 + lr][s*32], hi);
        cs[nd] = wmma_bf16(aq[s], bk, cs[nd]);
      }
    }
    #pragma unroll
    for (int nd = 0; nd < 4; ++nd) {
      int col = kt*64 + nd*16 + lr;
      #pragma unroll
      for (int v = 0; v < 8; ++v) {
        int row = q0 + w*16 + v + 8*hi;
        float sc = cs[nd][v] * scale;
        cs[nd][v] = (col > row) ? -__builtin_inff() : sc;
      }
    }
    #pragma unroll
    for (int v = 0; v < 8; ++v) {
      float t = fmaxf(fmaxf(cs[0][v], cs[1][v]), fmaxf(cs[2][v], cs[3][v]));
      t = redmax16(t);
      float mnew = fmaxf(mrow[v], t);
      float su = __expf(cs[0][v]-mnew) + __expf(cs[1][v]-mnew)
               + __expf(cs[2][v]-mnew) + __expf(cs[3][v]-mnew);
      su = redsum16(su);
      lrow[v] = lrow[v] * __expf(mrow[v] - mnew) + su;
      mrow[v] = mnew;
    }
  }
  float invl[8];
  #pragma unroll
  for (int v = 0; v < 8; ++v) invl[v] = 1.0f / lrow[v];

  // ---- pass 2: write attn, accumulate out = P.V ----
  v8f co[4] = {};
  for (int kt = 0; kt <= qt; ++kt) {
    __syncthreads();
    for (int i = tid; i < 512; i += 128) {
      int r = i >> 3, cch = i & 7;
      async_copy16(kb + (rowbase + (size_t)kt*64 + r) * DMODEL + d0 + cch*8,
                   &Kl[r][cch*8]);
      async_copy16(vtb + vt_head + (size_t)r * SEQ + kt*64 + cch*8,
                   &Vt[r][cch*8]);
    }
    wait_async();
    __syncthreads();

    v8f cs[4] = {};
    #pragma unroll
    for (int s = 0; s < 2; ++s) {
      #pragma unroll
      for (int nd = 0; nd < 4; ++nd) {
        v16bf bk = frag_b(&Kl[nd*16 + lr][s*32], hi);
        cs[nd] = wmma_bf16(aq[s], bk, cs[nd]);
      }
    }
    size_t abase = ((size_t)(b*HEADS + h) * SEQ) * SEQ;
    #pragma unroll
    for (int nd = 0; nd < 4; ++nd) {
      int coll = nd*16 + lr;
      int col  = kt*64 + coll;
      #pragma unroll
      for (int v = 0; v < 8; ++v) {
        int rloc = w*16 + v + 8*hi;
        int row  = q0 + rloc;
        float sc = cs[nd][v] * scale;
        sc = (col > row) ? -__builtin_inff() : sc;
        float p = __expf(sc - mrow[v]) * invl[v];     // masked -> 0
        attnp[abase + (size_t)row * SEQ + col] = p;
        Pl[w][v + 8*hi][coll] = (__bf16)p;
      }
    }
    // O += P . V  (per-wave P buffer; compiler inserts ds waits)
    #pragma unroll
    for (int s = 0; s < 2; ++s) {
      v16bf ap = frag_a(&Pl[w][lr][s*32], hi);
      #pragma unroll
      for (int nd = 0; nd < 4; ++nd) {
        v16bf bv2 = frag_b(&Vt[nd*16 + lr][s*32], hi);
        co[nd] = wmma_bf16(ap, bv2, co[nd]);
      }
    }
  }

  // ---- write out tile ----
  #pragma unroll
  for (int nd = 0; nd < 4; ++nd) {
    #pragma unroll
    for (int v = 0; v < 8; ++v) {
      int row = q0 + w*16 + v + 8*hi;
      outp[(rowbase + row) * DMODEL + d0 + nd*16 + lr] = co[nd][v];
    }
  }

  // ---- zero-fill strictly-upper causal tiles of attn for these 64 rows ----
  int c0z = (qt + 1) * 64;
  int colsZ = SEQ - c0z;
  if (colsZ > 0) {
    size_t base = ((size_t)(b*HEADS + h) * SEQ + q0) * SEQ;
    int n4 = colsZ >> 2;
    for (int i = tid; i < 64 * n4; i += 128) {
      int r = i / n4, cq = i - r * n4;
      float4 zz; zz.x = 0.f; zz.y = 0.f; zz.z = 0.f; zz.w = 0.f;
      *(float4*)(attnp + base + (size_t)r * SEQ + c0z + cq*4) = zz;
    }
  }
}

extern "C" void kernel_launch(void* const* d_in, const int* in_sizes, int n_in,
                              void* d_out, int out_size, void* d_ws, size_t ws_size,
                              hipStream_t stream) {
  const float* x  = (const float*)d_in[0];
  const float* Wq = (const float*)d_in[1];
  const float* bq = (const float*)d_in[2];
  const float* Wk = (const float*)d_in[3];
  const float* bk = (const float*)d_in[4];
  const float* Wv = (const float*)d_in[5];
  const float* bv = (const float*)d_in[6];

  float* out  = (float*)d_out;
  float* attn = out + ATTN_OFF;
  float* Kb   = out + K_OFF;
  float* Vb   = out + V_OFF;

  // bf16 staging in workspace: Q | K | V^T(per-head)  (24 MB total)
  __bf16* qb  = (__bf16*)d_ws;
  __bf16* kbp = qb  + (size_t)MTOT * DMODEL;
  __bf16* vtb = kbp + (size_t)MTOT * DMODEL;

  dim3 g1(MTOT/64, DMODEL/64, 3);
  qkv_gemm_kernel<<<g1, 128, 0, stream>>>(x, Wq, bq, Wk, bk, Wv, bv,
                                          Kb, Vb, qb, kbp, vtb);

  dim3 g2(SEQ/64, HEADS, BATCH);
  attention_kernel<<<g2, 128, 0, stream>>>(qb, kbp, vtb, out, attn);
}
